// LSTMModel_32779190403176
// MI455X (gfx1250) — compile-verified, matrix operand order checked by reading, not run on hardware
//
#include <hip/hip_runtime.h>
#include <hip/hip_bf16.h>
#include <math.h>

typedef __attribute__((ext_vector_type(16))) __bf16 v16bf;
typedef __attribute__((ext_vector_type(8)))  float  v8f;

#define BB 64
#define TT 512
#define EE 256
#define UU 1024
#define VV 32000

union FragU { uint4 u[2]; v16bf v; };

// ---------------------------------------------------------------------------
// Pack W = [W0 ; W1] (f32, row-major KxN) into bf16 WMMA B-fragments.
// Fragment f = nt*kTiles + kt is 512 bf16 (1024 B), 32 B per lane, matching
// the 16-bit 32x16 B-matrix layout (lane n = lane&15; lane<16 holds K
// {s..s+7, s+16..s+23}, s = 0/8 by half-wave; consecutive-K pairs per dword).
// ---------------------------------------------------------------------------
__global__ void pack_b_kernel(const float* __restrict__ W0, int K0,
                              const float* __restrict__ W1,
                              int N, int kTiles, __bf16* __restrict__ out)
{
  int lane = threadIdx.x & 31;
  int f = blockIdx.x * (blockDim.x >> 5) + (threadIdx.x >> 5);
  int nt = f / kTiles;
  int kt = f - nt * kTiles;
  int n  = nt * 16 + (lane & 15);
  int s  = (lane < 16) ? 0 : 8;
  __bf16* o = out + (size_t)f * 512 + lane * 16;
  #pragma unroll
  for (int j = 0; j < 8; ++j) {
    int kl = (j < 4) ? (s + 2 * j) : (s + 16 + 2 * (j - 4));
    int k  = kt * 32 + kl;
    #pragma unroll
    for (int e = 0; e < 2; ++e) {
      int kk  = k + e;
      float v = (kk < K0) ? W0[(size_t)kk * N + n]
                          : W1[(size_t)(kk - K0) * N + n];
      o[2 * j + e] = (__bf16)v;
    }
  }
}

// 4-tile WMMA inner step: one A fragment feeds 4 independent WMMA chains.
#define WMMA_STEP4(AP)                                                         \
  {                                                                            \
    FragU a, b;                                                                \
    a.u[0] = *(const uint4*)(AP);                                              \
    a.u[1] = *(const uint4*)((AP) + 16);                                       \
    b.u[0] = b0p[0]; b.u[1] = b0p[1]; b0p += 64;                               \
    __builtin_prefetch(b0p, 0, 1);                                             \
    acc0 = __builtin_amdgcn_wmma_f32_16x16x32_bf16(false, a.v, false, b.v,     \
                                                   (short)0, acc0, false, false); \
    b.u[0] = b1p[0]; b.u[1] = b1p[1]; b1p += 64;                               \
    __builtin_prefetch(b1p, 0, 1);                                             \
    acc1 = __builtin_amdgcn_wmma_f32_16x16x32_bf16(false, a.v, false, b.v,     \
                                                   (short)0, acc1, false, false); \
    b.u[0] = b2p[0]; b.u[1] = b2p[1]; b2p += 64;                               \
    __builtin_prefetch(b2p, 0, 1);                                             \
    acc2 = __builtin_amdgcn_wmma_f32_16x16x32_bf16(false, a.v, false, b.v,     \
                                                   (short)0, acc2, false, false); \
    b.u[0] = b3p[0]; b.u[1] = b3p[1]; b3p += 64;                               \
    __builtin_prefetch(b3p, 0, 1);                                             \
    acc3 = __builtin_amdgcn_wmma_f32_16x16x32_bf16(false, a.v, false, b.v,     \
                                                   (short)0, acc3, false, false); \
  }

// ---------------------------------------------------------------------------
// Generic GEMM: C(M x N) = [A0 | A1] * Bp + bias, bf16 in, f32 accumulate.
// Wave owns one 16-row M-tile x four consecutive 16-col N-tiles.
// grid = (N/64, M/64), block = 128 (4 waves = 4 M-tiles).
// biasv (nullable): per-column f32 bias. Writes bf16 to Cb if non-null,
// else f32 to Cf.
// ---------------------------------------------------------------------------
__global__ void gemm_wmma_kernel(const __bf16* __restrict__ A0, int ldA0, int K0,
                                 const __bf16* __restrict__ A1, int ldA1,
                                 const __bf16* __restrict__ Bp,
                                 const float* __restrict__ biasv,
                                 float* __restrict__ Cf, __bf16* __restrict__ Cb,
                                 int N, int kTiles)
{
  int lane  = threadIdx.x & 31;
  int mt    = threadIdx.x >> 5;
  int mbase = blockIdx.y * 64 + mt * 16;
  int nt0   = blockIdx.x * 4;
  int row   = mbase + (lane & 15);
  int s     = (lane < 16) ? 0 : 8;
  int kT0   = K0 >> 5;

  const uint4* b0p = (const uint4*)(Bp + (size_t)(nt0 + 0) * kTiles * 512) + lane * 2;
  const uint4* b1p = (const uint4*)(Bp + (size_t)(nt0 + 1) * kTiles * 512) + lane * 2;
  const uint4* b2p = (const uint4*)(Bp + (size_t)(nt0 + 2) * kTiles * 512) + lane * 2;
  const uint4* b3p = (const uint4*)(Bp + (size_t)(nt0 + 3) * kTiles * 512) + lane * 2;

  v8f acc0 = {}, acc1 = {}, acc2 = {}, acc3 = {};

  const __bf16* ap = A0 + (size_t)row * ldA0 + s;
  for (int kt = 0; kt < kT0; ++kt) {
    WMMA_STEP4(ap);
    ap += 32;
  }
  if (kT0 < kTiles) {
    ap = A1 + (size_t)row * ldA1 + s;
    for (int kt = kT0; kt < kTiles; ++kt) {
      WMMA_STEP4(ap);
      ap += 32;
    }
  }

  int col0 = nt0 * 16 + (lane & 15);
  int mrow = mbase + ((lane < 16) ? 0 : 8);
  float bv0 = 0.f, bv1 = 0.f, bv2 = 0.f, bv3 = 0.f;
  if (biasv) {
    bv0 = biasv[col0];
    bv1 = biasv[col0 + 16];
    bv2 = biasv[col0 + 32];
    bv3 = biasv[col0 + 48];
  }
  if (Cb) {
    #pragma unroll
    for (int j = 0; j < 8; ++j) {
      __bf16* cr = Cb + (size_t)(mrow + j) * N + col0;
      cr[0]  = (__bf16)(acc0[j] + bv0);
      cr[16] = (__bf16)(acc1[j] + bv1);
      cr[32] = (__bf16)(acc2[j] + bv2);
      cr[48] = (__bf16)(acc3[j] + bv3);
    }
  } else {
    #pragma unroll
    for (int j = 0; j < 8; ++j) {
      float* cr = Cf + (size_t)(mrow + j) * N + col0;
      cr[0]  = acc0[j] + bv0;
      cr[16] = acc1[j] + bv1;
      cr[32] = acc2[j] + bv2;
      cr[48] = acc3[j] + bv3;
    }
  }
}

// ---------------------------------------------------------------------------
// Fused LSTM recurrence step: z = h_{t-1}@Wr (WMMA) + gates + state update.
// Block j (of 64) computes units u in [16j, 16j+16) for ALL FOUR gate
// quarters: B N-tiles {j, j+64, j+128, j+192} -> acc0..acc3 are exactly the
// i/f/g/o pre-activations for a 64-row x 16-unit patch. The epilogue adds
// the precomputed bf16 (x_t@Wk + bias), applies sigmoid/tanh, and does the
// masked c/h update in-register. One kernel per timestep.
// grid = (64,1), block = 128 (4 waves = 4 M-tiles of the 64 batch rows).
// ---------------------------------------------------------------------------
__global__ void lstm_step_kernel(const __bf16* __restrict__ Hp,   // h_{t-1}, 64xU bf16
                                 const __bf16* __restrict__ Bp,   // packed Wr
                                 const __bf16* __restrict__ xpz,  // x_t@Wk + b, 64x4U bf16
                                 float* __restrict__ c,           // 64xU f32 state
                                 __bf16* __restrict__ hout,       // 64xU bf16
                                 const int* __restrict__ x, int t)
{
  const int kTiles = UU / 32;                // 32
  int lane  = threadIdx.x & 31;
  int mt    = threadIdx.x >> 5;
  int mbase = mt * 16;
  int j     = blockIdx.x;                    // unit tile 0..63
  int row   = mbase + (lane & 15);
  int s     = (lane < 16) ? 0 : 8;

  const uint4* b0p = (const uint4*)(Bp + (size_t)(j)       * kTiles * 512) + lane * 2;
  const uint4* b1p = (const uint4*)(Bp + (size_t)(j + 64)  * kTiles * 512) + lane * 2;
  const uint4* b2p = (const uint4*)(Bp + (size_t)(j + 128) * kTiles * 512) + lane * 2;
  const uint4* b3p = (const uint4*)(Bp + (size_t)(j + 192) * kTiles * 512) + lane * 2;

  v8f acc0 = {}, acc1 = {}, acc2 = {}, acc3 = {};

  const __bf16* ap = Hp + (size_t)row * UU + s;
  for (int kt = 0; kt < kTiles; ++kt) {
    WMMA_STEP4(ap);
    ap += 32;
  }

  int u    = j * 16 + (lane & 15);           // unit index 0..1023
  int mrow = mbase + ((lane < 16) ? 0 : 8);
  #pragma unroll
  for (int jj = 0; jj < 8; ++jj) {
    int r = mrow + jj;                       // batch row
    const __bf16* xr = xpz + (size_t)r * 4 * UU;
    float zi = acc0[jj] + (float)xr[u];
    float zf = acc1[jj] + (float)xr[u + UU];
    float zg = acc2[jj] + (float)xr[u + 2 * UU];
    float zo = acc3[jj] + (float)xr[u + 3 * UU];
    float ig = 1.0f / (1.0f + __expf(-zi));
    float fg = 1.0f / (1.0f + __expf(-zf));
    float gg = tanhf(zg);
    float og = 1.0f / (1.0f + __expf(-zo));
    int   ci = r * UU + u;
    float cold = c[ci];
    float cn = fg * cold + ig * gg;
    float hn = og * tanhf(cn);
    bool  m  = (x[r * TT + t] != 0);
    c[ci]    = m ? cn : cold;
    hout[ci] = m ? (__bf16)hn : Hp[ci];
  }
}

// ---------------------------------------------------------------------------
// xs[t][r][e] = bf16(embed[x[r][t]][e])   (T,B,E layout, bf16)
// ---------------------------------------------------------------------------
__global__ void embed_kernel(const int* __restrict__ x,
                             const float* __restrict__ emb,
                             __bf16* __restrict__ out)
{
  int i = blockIdx.x * blockDim.x + threadIdx.x;   // T*B*E threads
  int e = i % EE;
  int r = (i / EE) % BB;
  int t = i / (EE * BB);
  int tok = x[r * TT + t];
  out[i] = (__bf16)emb[(size_t)tok * EE + e];
}

__global__ void zero_f32_kernel(float* __restrict__ p)
{
  p[blockIdx.x * blockDim.x + threadIdx.x] = 0.0f;
}

__global__ void zero_bf16_kernel(__bf16* __restrict__ p)
{
  p[blockIdx.x * blockDim.x + threadIdx.x] = (__bf16)0.0f;
}

// softmax over V=32000 per row (bias already folded into logits).
__global__ void softmax_kernel(const float* __restrict__ logits,
                               float* __restrict__ out)
{
  __shared__ float red[256];
  int r   = blockIdx.x;
  int tid = threadIdx.x;
  const float* lr = logits + (size_t)r * VV;
  float* orow     = out    + (size_t)r * VV;

  float mx = -3.4e38f;
  for (int v = tid; v < VV; v += 256) mx = fmaxf(mx, lr[v]);
  red[tid] = mx; __syncthreads();
  for (int st = 128; st > 0; st >>= 1) {
    if (tid < st) red[tid] = fmaxf(red[tid], red[tid + st]);
    __syncthreads();
  }
  mx = red[0]; __syncthreads();

  float sum = 0.0f;
  for (int v = tid; v < VV; v += 256) sum += __expf(lr[v] - mx);
  red[tid] = sum; __syncthreads();
  for (int st = 128; st > 0; st >>= 1) {
    if (tid < st) red[tid] += red[tid + st];
    __syncthreads();
  }
  float inv = 1.0f / red[0];
  for (int v = tid; v < VV; v += 256) orow[v] = __expf(lr[v] - mx) * inv;
}

// ---------------------------------------------------------------------------
extern "C" void kernel_launch(void* const* d_in, const int* in_sizes, int n_in,
                              void* d_out, int out_size, void* d_ws, size_t ws_size,
                              hipStream_t stream)
{
  (void)in_sizes; (void)n_in; (void)out_size; (void)ws_size;

  const int*   x   = (const int*)d_in[0];
  const float* emb = (const float*)d_in[1];
  const float* Wk[5], *Wr[5], *bb[5];
  for (int l = 0; l < 5; ++l) {
    Wk[l] = (const float*)d_in[2 + 3 * l];
    Wr[l] = (const float*)d_in[3 + 3 * l];
    bb[l] = (const float*)d_in[4 + 3 * l];
  }
  const float* hW1 = (const float*)d_in[17]; const float* hb1 = (const float*)d_in[18];
  const float* hW2 = (const float*)d_in[19]; const float* hb2 = (const float*)d_in[20];
  const float* hW3 = (const float*)d_in[21]; const float* hb3 = (const float*)d_in[22];

  char* ws = (char*)d_ws;
  size_t off = 0;
  auto alloc = [&](size_t bytes) -> void* {
    void* p = ws + off;
    off = (off + bytes + 255) & ~(size_t)255;
    return p;
  };

  // packed bf16 weights (fragment order)
  __bf16* packedWk[5];
  __bf16* packedWr[5];
  int kTk[5];
  for (int l = 0; l < 5; ++l) {
    int K0 = (l == 0) ? EE : UU;
    kTk[l] = K0 / 32;                                    // 8 or 32
    packedWk[l] = (__bf16*)alloc((size_t)kTk[l] * 256 * 1024);
    packedWr[l] = (__bf16*)alloc((size_t)32 * 256 * 1024);
  }
  __bf16* packedH1 = (__bf16*)alloc((size_t)32 * 64 * 1024);
  __bf16* packedH2 = (__bf16*)alloc((size_t)32 * 64 * 1024);
  __bf16* packedH3 = (__bf16*)alloc((size_t)32 * 2000 * 1024);

  __bf16* xemb  = (__bf16*)alloc((size_t)TT * BB * EE * 2);
  __bf16* hsA   = (__bf16*)alloc((size_t)TT * BB * UU * 2);
  __bf16* hsB   = (__bf16*)alloc((size_t)TT * BB * UU * 2);
  __bf16* xproj = (__bf16*)alloc((size_t)TT * BB * 4 * UU * 2);   // 256 MB
  float*  cbuf  = (float*)alloc((size_t)BB * UU * 4);
  __bf16* h0z   = (__bf16*)alloc((size_t)BB * UU * 2);
  __bf16* a1    = (__bf16*)alloc((size_t)BB * UU * 2);
  __bf16* a2    = (__bf16*)alloc((size_t)BB * UU * 2);
  float*  lg    = (float*)alloc((size_t)BB * VV * 4);

  // ---- pack weights (bf16, WMMA fragment order; resident in 192MB L2) ----
  for (int l = 0; l < 5; ++l) {
    int K0 = (l == 0) ? EE : UU;
    pack_b_kernel<<<(kTk[l] * 256) / 8, 256, 0, stream>>>(Wk[l], K0, Wk[l],
                                                          4 * UU, kTk[l],
                                                          packedWk[l]);
    pack_b_kernel<<<(32 * 256) / 8, 256, 0, stream>>>(Wr[l], UU, Wr[l],
                                                      4 * UU, 32, packedWr[l]);
  }
  pack_b_kernel<<<(32 * 64) / 8,   256, 0, stream>>>(hW1, UU, hW1, UU, 32, packedH1);
  pack_b_kernel<<<(32 * 64) / 8,   256, 0, stream>>>(hW2, UU, hW2, UU, 32, packedH2);
  pack_b_kernel<<<(32 * 2000) / 8, 256, 0, stream>>>(hW3, UU, hW3, VV, 32, packedH3);

  // ---- embedding gather -> (T,B,E) bf16 ----
  embed_kernel<<<(TT * BB * EE) / 256, 256, 0, stream>>>(x, emb, xemb);
  zero_bf16_kernel<<<(BB * UU) / 256, 256, 0, stream>>>(h0z);

  // ---- 5 stacked LSTMs ----
  for (int l = 0; l < 5; ++l) {
    int d_in = (l == 0) ? EE : UU;
    const __bf16* xin = (l == 0) ? xemb : ((l & 1) ? hsA : hsB);
    __bf16* hout      = (l & 1) ? hsB : hsA;

    // batched input projection over ALL timesteps: xproj = Xin @ Wk + b (bf16)
    {
      dim3 grid(4 * UU / 64, TT * BB / 64);
      gemm_wmma_kernel<<<grid, 128, 0, stream>>>(xin, d_in, d_in, xin, d_in,
                                                 packedWk[l], bb[l],
                                                 nullptr, xproj,
                                                 4 * UU, kTk[l]);
    }

    zero_f32_kernel<<<(BB * UU) / 256, 256, 0, stream>>>(cbuf);

    // fused sequential recurrence: one kernel per step (h@Wr + gates + mask)
    for (int t = 0; t < TT; ++t) {
      const __bf16* Hp = (t == 0) ? h0z : hout + (size_t)(t - 1) * BB * UU;
      lstm_step_kernel<<<64, 128, 0, stream>>>(
          Hp, packedWr[l], xproj + (size_t)t * BB * 4 * UU, cbuf,
          hout + (size_t)t * BB * UU, x, t);
    }
  }

  // final layer (l=4) wrote hsA; h_last = output at t = T-1
  const __bf16* hLast = hsA + (size_t)(TT - 1) * BB * UU;

  // ---- MLP head: biased GEMMs straight through, then softmax ----
  gemm_wmma_kernel<<<dim3(UU / 64, 1), 128, 0, stream>>>(
      hLast, UU, UU, hLast, UU, packedH1, hb1, nullptr, a1, UU, 32);
  gemm_wmma_kernel<<<dim3(UU / 64, 1), 128, 0, stream>>>(
      a1, UU, UU, a1, UU, packedH2, hb2, nullptr, a2, UU, 32);
  gemm_wmma_kernel<<<dim3(VV / 64, 1), 128, 0, stream>>>(
      a2, UU, UU, a2, UU, packedH3, hb3, lg, nullptr, VV, 32);
  softmax_kernel<<<64, 256, 0, stream>>>(lg, (float*)d_out);
}
